// MultiHeadAttention_3109556322588
// MI455X (gfx1250) — matrix operand order repaired and checked by reading, not compiled, not run on hardware
//
#include <hip/hip_runtime.h>
#include <hip/hip_bf16.h>

// ---------------------------------------------------------------------------
// MHA forward for MI455X (gfx1250, wave32, WMMA 16x16x32 bf16,
// async global->LDS DMA with double buffering)
// ---------------------------------------------------------------------------

typedef __attribute__((ext_vector_type(8)))  float  v8f;
typedef __attribute__((ext_vector_type(16))) __bf16 v16bf;

#define EMBED_DIM 1024
#define NUM_HEADS 16
#define HEAD_DIM  64
#define BATCH     2
#define SEQ       2048

#define BM  128
#define BN  128
#define BK  64
#define LDT 72   // padded LDS row stride (elements): 144B -> conflict-free, 16B aligned

union Frag {
    v16bf v;
    uint4 q[2];
};

__device__ __forceinline__ v8f v8f_zero() {
    v8f z = {0.f,0.f,0.f,0.f,0.f,0.f,0.f,0.f};
    return z;
}

// xor-shuffle within the 32-lane wave (stays inside each 16-lane half for xm<=8)
__device__ __forceinline__ float bperm_xor(float v, int xm) {
    int lane = (int)(threadIdx.x & 31u);
    int idx  = (lane ^ xm) << 2;
    return __int_as_float(__builtin_amdgcn_ds_bpermute(idx, __float_as_int(v)));
}

// Async DMA: copy 16 bytes per lane from global memory straight into LDS.
// Tracked by ASYNCcnt (CDNA5).  LDS offset = low 32 bits of the flat address
// of a __shared__ object (shared aperture keeps the LDS offset in addr[31:0]).
__device__ __forceinline__ void async_copy_b128(void* lds_ptr, const void* gptr) {
    uint32_t l = (uint32_t)(uintptr_t)lds_ptr;
    asm volatile("global_load_async_to_lds_b128 %0, %1, off"
                 :: "v"(l), "v"(gptr)
                 : "memory");
}
__device__ __forceinline__ void wait_async0() {
    asm volatile("s_wait_asynccnt 0" ::: "memory");
}

// ---------------------------------------------------------------------------
// fp32 -> bf16 conversion (vectorized x4)
// ---------------------------------------------------------------------------
__global__ void cvt_f32_bf16(const float* __restrict__ in,
                             __bf16* __restrict__ out, int n) {
    int i = (int)(blockIdx.x * blockDim.x + threadIdx.x) * 4;
    if (i + 3 < n) {
        float4 f = *reinterpret_cast<const float4*>(in + i);
        union { uint2 u; __bf16 h[4]; } b;
        b.h[0] = (__bf16)f.x; b.h[1] = (__bf16)f.y;
        b.h[2] = (__bf16)f.z; b.h[3] = (__bf16)f.w;
        *reinterpret_cast<uint2*>(out + i) = b.u;
    }
}

// ---------------------------------------------------------------------------
// GEMM: Y[M,N] = X[M,K] * W[N,K]^T + bias[N]
// Block tile 128x128, 8 waves (2x4), wave tile 64x32 (4x2 WMMA tiles).
// K staged 64-deep in double-buffered LDS filled by async DMA.
// outb != null -> bf16 row-major output; else outf (fp32 row-major).
// ---------------------------------------------------------------------------
__global__ __launch_bounds__(256)
void gemm_bf16_wmma(const __bf16* __restrict__ X, const __bf16* __restrict__ W,
                    const float* __restrict__ bias,
                    __bf16* __restrict__ outb, float* __restrict__ outf,
                    int M, int N, int K) {
    __shared__ __bf16 As[2][BM * LDT];
    __shared__ __bf16 Bs[2][BN * LDT];

    const int tid  = threadIdx.x;
    const int lane = tid & 31;
    const int wave = tid >> 5;
    const int l16  = lane & 15;
    const int kh   = (lane >> 4) & 1;
    const int wm   = wave >> 2;                // 0..1  (64 rows each)
    const int wn   = wave & 3;                 // 0..3  (32 cols each)
    const int bm   = blockIdx.y * BM;
    const int bn   = blockIdx.x * BN;

    v8f acc[4][2];
    #pragma unroll
    for (int mt = 0; mt < 4; ++mt)
        #pragma unroll
        for (int nt = 0; nt < 2; ++nt)
            acc[mt][nt] = v8f_zero();

    // issue async DMA for one 128x64 tile of X and W into buffer `buf`
    auto stage = [&](int buf, int k0) {
        #pragma unroll
        for (int c = 0; c < 4; ++c) {
            int chunk = tid + c * 256;         // 0..1023
            int r  = chunk >> 3;               // 0..127
            int kq = (chunk & 7) << 3;         // 0..56 step 8
            async_copy_b128(&As[buf][r * LDT + kq],
                            X + (size_t)(bm + r) * K + k0 + kq);
            async_copy_b128(&Bs[buf][r * LDT + kq],
                            W + (size_t)(bn + r) * K + k0 + kq);
        }
    };

    stage(0, 0);
    wait_async0();
    __syncthreads();

    int cur = 0;
    for (int k0 = 0; k0 < K; k0 += BK) {
        if (k0 + BK < K) {
            stage(cur ^ 1, k0 + BK);           // prefetch next tile via DMA
            __builtin_prefetch(X + (size_t)(bm + (tid >> 1)) * K + k0 + BK, 0, 1);
        }

        #pragma unroll
        for (int ks = 0; ks < BK; ks += 32) {
            Frag af[4], bf[2];
            #pragma unroll
            for (int mt = 0; mt < 4; ++mt) {
                const __bf16* p = &As[cur][(wm*64 + mt*16 + l16) * LDT + ks + kh*8];
                af[mt].q[0] = *reinterpret_cast<const uint4*>(p);
                af[mt].q[1] = *reinterpret_cast<const uint4*>(p + 16);
            }
            #pragma unroll
            for (int nt = 0; nt < 2; ++nt) {
                const __bf16* p = &Bs[cur][(wn*32 + nt*16 + l16) * LDT + ks + kh*8];
                bf[nt].q[0] = *reinterpret_cast<const uint4*>(p);
                bf[nt].q[1] = *reinterpret_cast<const uint4*>(p + 16);
            }
            #pragma unroll
            for (int mt = 0; mt < 4; ++mt)
                #pragma unroll
                for (int nt = 0; nt < 2; ++nt)
                    acc[mt][nt] = __builtin_amdgcn_wmma_f32_16x16x32_bf16(
                        false, af[mt].v, false, bf[nt].v,
                        (short)0, acc[mt][nt], false, false);
        }

        wait_async0();       // next tile landed in the alternate buffer
        __syncthreads();     // all waves done reading the current buffer
        cur ^= 1;
    }

    // epilogue: bias + store. C layout: lane col = l16, row = vi + 8*kh.
    #pragma unroll
    for (int nt = 0; nt < 2; ++nt) {
        int col = bn + wn*32 + nt*16 + l16;
        float bv = bias[col];
        #pragma unroll
        for (int mt = 0; mt < 4; ++mt) {
            #pragma unroll
            for (int vi = 0; vi < 8; ++vi) {
                int row = bm + wm*64 + mt*16 + vi + kh*8;
                float val = acc[mt][nt][vi] + bv;
                if (outf) outf[(size_t)row * N + col] = val;
                else      outb[(size_t)row * N + col] = (__bf16)val;
            }
        }
    }
}

// ---------------------------------------------------------------------------
// Flash attention: one workgroup per (b, h, 128-row Q tile).
// Q/K/V stored bf16 [B, S, D] with head slice at column h*64.
// Each wave owns 16 Q rows; online softmax stats per row in registers.
// ---------------------------------------------------------------------------
__global__ __launch_bounds__(256)
void attn_wmma(const __bf16* __restrict__ Qp, const __bf16* __restrict__ Kp,
               const __bf16* __restrict__ Vp, __bf16* __restrict__ O) {
    __shared__ __bf16 Qs[128 * LDT];
    __shared__ __bf16 Ks[64 * LDT];
    __shared__ __bf16 Vt[64 * LDT];       // transposed: row = head-dim, col = key
    __shared__ __bf16 Ps[8 * 16 * LDT];   // per-wave P tile for layout fixup

    const int tid  = threadIdx.x;
    const int lane = tid & 31;
    const int wave = tid >> 5;
    const int l16  = lane & 15;
    const int kh   = (lane >> 4) & 1;
    const int q0   = blockIdx.x * 128;
    const int b    = blockIdx.y / NUM_HEADS;
    const int h    = blockIdx.y % NUM_HEADS;
    const size_t base = (size_t)b * SEQ * EMBED_DIM + (size_t)h * HEAD_DIM;

    // stage Q tile (128 x 64) via async DMA
    #pragma unroll
    for (int c = 0; c < 4; ++c) {
        int chunk = tid + c * 256;
        int r  = chunk >> 3;
        int kq = (chunk & 7) << 3;
        async_copy_b128(&Qs[r * LDT + kq],
                        Qp + base + (size_t)(q0 + r) * EMBED_DIM + kq);
    }
    wait_async0();
    __syncthreads();

    Frag qf[2];
    #pragma unroll
    for (int kk = 0; kk < 2; ++kk) {
        const __bf16* p = &Qs[(wave*16 + l16) * LDT + kk*32 + kh*8];
        qf[kk].q[0] = *reinterpret_cast<const uint4*>(p);
        qf[kk].q[1] = *reinterpret_cast<const uint4*>(p + 16);
    }

    v8f o[4];
    float mrow[8], lrow[8];
    #pragma unroll
    for (int nt = 0; nt < 4; ++nt) o[nt] = v8f_zero();
    #pragma unroll
    for (int vi = 0; vi < 8; ++vi) { mrow[vi] = -3.0e38f; lrow[vi] = 0.f; }

    for (int kc = 0; kc < SEQ; kc += 64) {
        __syncthreads();
        // stage K chunk (64x64, async DMA) and transposed V chunk (manual)
        #pragma unroll
        for (int c = 0; c < 2; ++c) {
            int chunk = tid + c * 256;
            int r  = chunk >> 3;
            int kq = (chunk & 7) << 3;
            async_copy_b128(&Ks[r * LDT + kq],
                            Kp + base + (size_t)(kc + r) * EMBED_DIM + kq);
            union { uint4 u; __bf16 hh[8]; } dv;
            dv.u = *reinterpret_cast<const uint4*>(
                Vp + base + (size_t)(kc + r) * EMBED_DIM + kq);
            #pragma unroll
            for (int j = 0; j < 8; ++j) Vt[(kq + j) * LDT + r] = dv.hh[j];
        }
        wait_async0();
        __syncthreads();

        // S = Q * K^T  (16x64 per wave), scale 1/sqrt(64)
        v8f sc[4];
        #pragma unroll
        for (int nt = 0; nt < 4; ++nt) {
            v8f s = v8f_zero();
            #pragma unroll
            for (int kk = 0; kk < 2; ++kk) {
                Frag kb;
                const __bf16* p = &Ks[(nt*16 + l16) * LDT + kk*32 + kh*8];
                kb.q[0] = *reinterpret_cast<const uint4*>(p);
                kb.q[1] = *reinterpret_cast<const uint4*>(p + 16);
                s = __builtin_amdgcn_wmma_f32_16x16x32_bf16(
                    false, qf[kk].v, false, kb.v, (short)0, s, false, false);
            }
            #pragma unroll
            for (int vi = 0; vi < 8; ++vi) s[vi] *= 0.125f;
            sc[nt] = s;
        }

        // online softmax update (rows split across lane halves: row = vi + 8*kh)
        #pragma unroll
        for (int vi = 0; vi < 8; ++vi) {
            float mc = fmaxf(fmaxf(sc[0][vi], sc[1][vi]),
                             fmaxf(sc[2][vi], sc[3][vi]));
            #pragma unroll
            for (int xm = 1; xm <= 8; xm <<= 1) mc = fmaxf(mc, bperm_xor(mc, xm));
            float mn   = fmaxf(mrow[vi], mc);
            float scal = __expf(mrow[vi] - mn);
            mrow[vi]   = mn;
            float rs = 0.f;
            #pragma unroll
            for (int nt = 0; nt < 4; ++nt) {
                float pexp = __expf(sc[nt][vi] - mn);
                rs += pexp;
                Ps[(wave*16 + vi + kh*8) * LDT + nt*16 + l16] = (__bf16)pexp;
            }
            #pragma unroll
            for (int xm = 1; xm <= 8; xm <<= 1) rs += bperm_xor(rs, xm);
            lrow[vi] = lrow[vi] * scal + rs;
            #pragma unroll
            for (int nt = 0; nt < 4; ++nt) o[nt][vi] *= scal;
        }
        __syncthreads();

        // O += P * V
        Frag pf[2];
        #pragma unroll
        for (int kk = 0; kk < 2; ++kk) {
            const __bf16* p = &Ps[(wave*16 + l16) * LDT + kk*32 + kh*8];
            pf[kk].q[0] = *reinterpret_cast<const uint4*>(p);
            pf[kk].q[1] = *reinterpret_cast<const uint4*>(p + 16);
        }
        #pragma unroll
        for (int nt = 0; nt < 4; ++nt) {
            #pragma unroll
            for (int kk = 0; kk < 2; ++kk) {
                Frag vb;
                const __bf16* p = &Vt[(nt*16 + l16) * LDT + kk*32 + kh*8];
                vb.q[0] = *reinterpret_cast<const uint4*>(p);
                vb.q[1] = *reinterpret_cast<const uint4*>(p + 16);
                o[nt] = __builtin_amdgcn_wmma_f32_16x16x32_bf16(
                    false, pf[kk].v, false, vb.v, (short)0, o[nt], false, false);
            }
        }
    }

    // normalize and store (bf16, [B, S, D] with head offset)
    #pragma unroll
    for (int vi = 0; vi < 8; ++vi) {
        float inv = 1.0f / lrow[vi];
        int row = q0 + wave*16 + vi + kh*8;
        size_t rb = (size_t)b * SEQ * EMBED_DIM + (size_t)row * EMBED_DIM
                  + (size_t)h * HEAD_DIM;
        #pragma unroll
        for (int nt = 0; nt < 4; ++nt)
            O[rb + nt*16 + l16] = (__bf16)(o[nt][vi] * inv);
    }
}

// ---------------------------------------------------------------------------
// Host-side launch
// ---------------------------------------------------------------------------
extern "C" void kernel_launch(void* const* d_in, const int* in_sizes, int n_in,
                              void* d_out, int out_size, void* d_ws, size_t ws_size,
                              hipStream_t stream) {
    (void)in_sizes; (void)n_in; (void)out_size; (void)ws_size;

    const float* q  = (const float*)d_in[0];
    const float* k  = (const float*)d_in[1];
    const float* v  = (const float*)d_in[2];
    const float* Wq = (const float*)d_in[3];
    const float* bq = (const float*)d_in[4];
    const float* Wk = (const float*)d_in[5];
    const float* bk = (const float*)d_in[6];
    const float* Wv = (const float*)d_in[7];
    const float* bv = (const float*)d_in[8];
    const float* Wo = (const float*)d_in[9];
    const float* bo = (const float*)d_in[10];

    const int M = BATCH * SEQ;           // 4096
    const int D = EMBED_DIM;             // 1024
    const size_t act = (size_t)M * D;    // 4 Mi elements
    const size_t wsz = (size_t)D * D;    // 1 Mi elements

    __bf16* qb  = (__bf16*)d_ws;
    __bf16* kb  = qb  + act;
    __bf16* vb  = kb  + act;
    __bf16* wqb = vb  + act;
    __bf16* wkb = wqb + wsz;
    __bf16* wvb = wkb + wsz;
    __bf16* wob = wvb + wsz;
    __bf16* Qp  = wob + wsz;
    __bf16* Kp  = Qp  + act;
    __bf16* Vp  = Kp  + act;
    __bf16* AO  = Vp  + act;             // total ~67 MB of workspace

    // fp32 -> bf16 conversions
    cvt_f32_bf16<<<(int)(act / 1024), 256, 0, stream>>>(q,  qb,  (int)act);
    cvt_f32_bf16<<<(int)(act / 1024), 256, 0, stream>>>(k,  kb,  (int)act);
    cvt_f32_bf16<<<(int)(act / 1024), 256, 0, stream>>>(v,  vb,  (int)act);
    cvt_f32_bf16<<<(int)(wsz / 1024), 256, 0, stream>>>(Wq, wqb, (int)wsz);
    cvt_f32_bf16<<<(int)(wsz / 1024), 256, 0, stream>>>(Wk, wkb, (int)wsz);
    cvt_f32_bf16<<<(int)(wsz / 1024), 256, 0, stream>>>(Wv, wvb, (int)wsz);
    cvt_f32_bf16<<<(int)(wsz / 1024), 256, 0, stream>>>(Wo, wob, (int)wsz);

    // Q/K/V projections (bf16 out)
    dim3 gg(D / BN, M / BM);             // (8, 32)
    gemm_bf16_wmma<<<gg, 256, 0, stream>>>(qb, wqb, bq, Qp, nullptr, M, D, D);
    gemm_bf16_wmma<<<gg, 256, 0, stream>>>(kb, wkb, bk, Kp, nullptr, M, D, D);
    gemm_bf16_wmma<<<gg, 256, 0, stream>>>(vb, wvb, bv, Vp, nullptr, M, D, D);

    // flash attention
    dim3 ga(SEQ / 128, BATCH * NUM_HEADS);   // (16, 32)
    attn_wmma<<<ga, 256, 0, stream>>>(Qp, Kp, Vp, AO);

    // output projection (fp32 out)
    gemm_bf16_wmma<<<gg, 256, 0, stream>>>(AO, wob, bo, nullptr, (float*)d_out, M, D, D);
}
